// Decoder_25718264168590
// MI455X (gfx1250) — matrix operand order repaired and checked by reading, not compiled, not run on hardware
//
#include <hip/hip_runtime.h>
#include <cstdint>
#include <cstddef>

// Problem dims (match reference)
constexpr int B = 32, T = 50, H = 1024, E = 128, V = 32000;
constexpr int CATW = H + E;   // 1152

typedef __attribute__((ext_vector_type(16))) __bf16       bf16x16;
typedef __attribute__((ext_vector_type(8)))  float        f32x8;
typedef __attribute__((ext_vector_type(4)))  unsigned int u32x4;

union FragBF {
    bf16x16 v;
    u32x4   q[2];
};

__device__ __forceinline__ unsigned short f32_to_bf16(float f) {
    unsigned int u = __float_as_uint(f);
    unsigned int r = u + 0x7FFFu + ((u >> 16) & 1u);   // round-to-nearest-even
    return (unsigned short)(r >> 16);
}

__device__ __forceinline__ float sigm(float x) {
    return 1.0f / (1.0f + __expf(-x));
}

__device__ __forceinline__ f32x8 wmma_bf16(const FragBF& a, const FragBF& b, f32x8 c) {
    return __builtin_amdgcn_wmma_f32_16x16x32_bf16(false, a.v, false, b.v,
                                                   (short)0, c, false, false);
}

// ---------------------------------------------------------------------------
// Elementwise conversion f32 -> bf16 bits
// ---------------------------------------------------------------------------
__global__ void k_f32_to_bf16(const float* __restrict__ in,
                              unsigned short* __restrict__ outp, size_t n) {
    size_t i = (size_t)blockIdx.x * 256 + threadIdx.x;
    if (i < n) outp[i] = f32_to_bf16(in[i]);
}

// Build concatenated input [B*T, H+E] in bf16: [target_t | enc]
__global__ void k_build_cat(const float* __restrict__ target,  // [B,T,E]
                            const float* __restrict__ enc,     // [B,H]
                            unsigned short* __restrict__ cat)  // [B*T, CATW]
{
    size_t i = (size_t)blockIdx.x * 256 + threadIdx.x;
    constexpr size_t total = (size_t)B * T * CATW;
    if (i >= total) return;
    int k  = (int)(i % CATW);
    int bt = (int)(i / CATW);
    int t  = bt % T;
    int b  = bt / T;
    float v = (k < E) ? target[((size_t)b * T + t) * E + k]
                      : enc[(size_t)b * H + (k - E)];
    cat[i] = f32_to_bf16(v);
}

// Initial states: h0 = h1 = bf16(enc); c0 = c1 = enc (f32)
__global__ void k_init_states(const float* __restrict__ enc,
                              unsigned short* __restrict__ h0,
                              unsigned short* __restrict__ h1,
                              float* __restrict__ c0,
                              float* __restrict__ c1)
{
    int i = blockIdx.x * 256 + threadIdx.x;
    if (i >= B * H) return;
    unsigned short q = f32_to_bf16(enc[i]);
    h0[i] = q; h1[i] = q;
    c0[i] = enc[i]; c1[i] = enc[i];
}

// ---------------------------------------------------------------------------
// Projection: inp = cat_t @ proj_w^T + proj_b  ([32 x 1152] x [1152 -> 128])
// grid = E/16 blocks, 32 threads (1 wave). Wave computes 32x16 tile.
// ---------------------------------------------------------------------------
__global__ void k_proj(const unsigned short* __restrict__ cat,  // [B*T, CATW]
                       int t,
                       const unsigned short* __restrict__ Pw,   // [E, CATW] bf16
                       const float* __restrict__ pb,            // [E]
                       unsigned short* __restrict__ inp)        // [B, E] bf16
{
    const int lane = threadIdx.x;
    const int sub  = lane >> 4;      // 0/1: lane half
    const int lm   = lane & 15;
    const int n0   = blockIdx.x * 16;

    f32x8 zero = {0.f,0.f,0.f,0.f,0.f,0.f,0.f,0.f};
    f32x8 acc[2] = {zero, zero};

    for (int k0 = 0; k0 < CATW; k0 += 32) {
        FragBF a[2], b;
#pragma unroll
        for (int mt = 0; mt < 2; ++mt) {
            int m = mt * 16 + lm;   // batch row
            const u32x4* p = (const u32x4*)(cat + ((size_t)m * T + t) * CATW + k0 + sub * 8);
            a[mt].q[0] = p[0];
            a[mt].q[1] = p[2];
        }
        {
            const u32x4* p = (const u32x4*)(Pw + (size_t)(n0 + lm) * CATW + k0 + sub * 8);
            b.q[0] = p[0];
            b.q[1] = p[2];
        }
#pragma unroll
        for (int mt = 0; mt < 2; ++mt) acc[mt] = wmma_bf16(a[mt], b, acc[mt]);
    }

    const int n = n0 + lm;
    const float bv = pb[n];
#pragma unroll
    for (int mt = 0; mt < 2; ++mt) {
#pragma unroll
        for (int r = 0; r < 8; ++r) {
            int m = mt * 16 + sub * 8 + r;
            inp[(size_t)m * E + n] = f32_to_bf16(acc[mt][r] + bv);
        }
    }
}

// ---------------------------------------------------------------------------
// LSTM cell: gates = x @ Wih^T + h @ Whh^T + b ; update c,h for 16 columns.
// grid = H/16 blocks, 32 threads (1 wave). 8 accumulators = 2 M-tiles x 4 gates.
// ---------------------------------------------------------------------------
__global__ void k_cell(const unsigned short* __restrict__ xA, int Kx,  // [B,Kx] bf16
                       const unsigned short* __restrict__ Wih,        // [4H,Kx] bf16
                       const unsigned short* __restrict__ hA,         // [B,H] bf16
                       const unsigned short* __restrict__ Whh,        // [4H,H] bf16
                       const float* __restrict__ bih,
                       const float* __restrict__ bhh,
                       float* __restrict__ c,                         // [B,H] in/out
                       unsigned short* __restrict__ hOut,             // [B,H] bf16
                       unsigned short* __restrict__ hHist,            // [B*T,H] or null
                       int t)
{
    const int lane = threadIdx.x;
    const int sub  = lane >> 4;
    const int lm   = lane & 15;
    const int n0   = blockIdx.x * 16;

    f32x8 zero = {0.f,0.f,0.f,0.f,0.f,0.f,0.f,0.f};
    f32x8 acc[2][4];
#pragma unroll
    for (int i = 0; i < 2; ++i)
#pragma unroll
        for (int j = 0; j < 4; ++j) acc[i][j] = zero;

    // x @ Wih^T
    for (int k0 = 0; k0 < Kx; k0 += 32) {
        FragBF a[2], b[4];
#pragma unroll
        for (int mt = 0; mt < 2; ++mt) {
            const u32x4* p = (const u32x4*)(xA + (size_t)(mt * 16 + lm) * Kx + k0 + sub * 8);
            a[mt].q[0] = p[0];
            a[mt].q[1] = p[2];
        }
#pragma unroll
        for (int g = 0; g < 4; ++g) {
            const u32x4* p = (const u32x4*)(Wih + (size_t)(g * H + n0 + lm) * Kx + k0 + sub * 8);
            b[g].q[0] = p[0];
            b[g].q[1] = p[2];
        }
#pragma unroll
        for (int g = 0; g < 4; ++g) {
            acc[0][g] = wmma_bf16(a[0], b[g], acc[0][g]);
            acc[1][g] = wmma_bf16(a[1], b[g], acc[1][g]);
        }
    }

    // h @ Whh^T  (K = H)
    for (int k0 = 0; k0 < H; k0 += 32) {
        FragBF a[2], b[4];
#pragma unroll
        for (int mt = 0; mt < 2; ++mt) {
            const u32x4* p = (const u32x4*)(hA + (size_t)(mt * 16 + lm) * H + k0 + sub * 8);
            a[mt].q[0] = p[0];
            a[mt].q[1] = p[2];
        }
#pragma unroll
        for (int g = 0; g < 4; ++g) {
            const u32x4* p = (const u32x4*)(Whh + (size_t)(g * H + n0 + lm) * H + k0 + sub * 8);
            b[g].q[0] = p[0];
            b[g].q[1] = p[2];
        }
#pragma unroll
        for (int g = 0; g < 4; ++g) {
            acc[0][g] = wmma_bf16(a[0], b[g], acc[0][g]);
            acc[1][g] = wmma_bf16(a[1], b[g], acc[1][g]);
        }
    }

    // Cell update epilogue
    const int n = n0 + lm;
    const float bi = bih[n]         + bhh[n];
    const float bf = bih[H + n]     + bhh[H + n];
    const float bg = bih[2 * H + n] + bhh[2 * H + n];
    const float bo = bih[3 * H + n] + bhh[3 * H + n];

#pragma unroll
    for (int mt = 0; mt < 2; ++mt) {
#pragma unroll
        for (int r = 0; r < 8; ++r) {
            const int m = mt * 16 + sub * 8 + r;
            const size_t idx = (size_t)m * H + n;
            const float gi = sigm(acc[mt][0][r] + bi);
            const float gf = sigm(acc[mt][1][r] + bf);
            const float gg = tanhf(acc[mt][2][r] + bg);
            const float go = sigm(acc[mt][3][r] + bo);
            const float cn = gf * c[idx] + gi * gg;
            const float hn = go * tanhf(cn);
            c[idx] = cn;
            const unsigned short hb = f32_to_bf16(hn);
            hOut[idx] = hb;
            if (hHist) hHist[((size_t)m * T + t) * H + n] = hb;
        }
    }
}

// ---------------------------------------------------------------------------
// Batched vocab GEMM: out[1600, 32000] = Hs[1600,1024] @ lin_w[32000,1024]^T + b
// block = 256 threads (8 waves), tile 64(M) x 256(N); wave: 2x4 16x16 accums.
// ---------------------------------------------------------------------------
__global__ void __launch_bounds__(256)
k_gemm_logits(const unsigned short* __restrict__ A,   // [B*T, H] bf16
              const unsigned short* __restrict__ W,   // [V, H]   bf16
              const float* __restrict__ bias,         // [V]
              float* __restrict__ out)                // [B*T, V]
{
    const int lane = threadIdx.x & 31;
    const int wv   = threadIdx.x >> 5;
    const int sub  = lane >> 4;
    const int lm   = lane & 15;
    const int wm   = wv & 1;     // 0..1
    const int wn   = wv >> 1;    // 0..3
    const int m0 = blockIdx.y * 64  + wm * 32;
    const int n0 = blockIdx.x * 256 + wn * 64;

    f32x8 zero = {0.f,0.f,0.f,0.f,0.f,0.f,0.f,0.f};
    f32x8 acc[2][4];
#pragma unroll
    for (int i = 0; i < 2; ++i)
#pragma unroll
        for (int j = 0; j < 4; ++j) acc[i][j] = zero;

    for (int k0 = 0; k0 < H; k0 += 32) {
        FragBF a[2], b[4];
#pragma unroll
        for (int mt = 0; mt < 2; ++mt) {
            const u32x4* p = (const u32x4*)(A + (size_t)(m0 + mt * 16 + lm) * H + k0 + sub * 8);
            a[mt].q[0] = p[0];
            a[mt].q[1] = p[2];
        }
#pragma unroll
        for (int nt = 0; nt < 4; ++nt) {
            const u32x4* p = (const u32x4*)(W + (size_t)(n0 + nt * 16 + lm) * H + k0 + sub * 8);
            b[nt].q[0] = p[0];
            b[nt].q[1] = p[2];
        }
#pragma unroll
        for (int mt = 0; mt < 2; ++mt)
#pragma unroll
            for (int nt = 0; nt < 4; ++nt)
                acc[mt][nt] = wmma_bf16(a[mt], b[nt], acc[mt][nt]);
    }

#pragma unroll
    for (int nt = 0; nt < 4; ++nt) {
        const int n = n0 + nt * 16 + lm;
        const float bv = bias[n];
#pragma unroll
        for (int mt = 0; mt < 2; ++mt) {
#pragma unroll
            for (int r = 0; r < 8; ++r) {
                const int m = m0 + mt * 16 + sub * 8 + r;
                out[(size_t)m * V + n] = acc[mt][nt][r] + bv;
            }
        }
    }
}

// ---------------------------------------------------------------------------
// In-place log-softmax per row of [B*T, V]
// ---------------------------------------------------------------------------
__global__ void k_logsoftmax(float* __restrict__ out) {
    __shared__ float red[256];
    float* row = out + (size_t)blockIdx.x * V;
    const int tid = threadIdx.x;

    float m = -3.4e38f;
    for (int i = tid; i < V; i += 256) m = fmaxf(m, row[i]);
    red[tid] = m;
    __syncthreads();
    for (int s = 128; s > 0; s >>= 1) {
        if (tid < s) red[tid] = fmaxf(red[tid], red[tid + s]);
        __syncthreads();
    }
    const float mx = red[0];
    __syncthreads();

    float sum = 0.f;
    for (int i = tid; i < V; i += 256) sum += __expf(row[i] - mx);
    red[tid] = sum;
    __syncthreads();
    for (int s = 128; s > 0; s >>= 1) {
        if (tid < s) red[tid] += red[tid + s];
        __syncthreads();
    }
    const float ls = __logf(red[0]) + mx;

    for (int i = tid; i < V; i += 256) row[i] = row[i] - ls;
}

// ---------------------------------------------------------------------------
// Host launcher
// ---------------------------------------------------------------------------
extern "C" void kernel_launch(void* const* d_in, const int* in_sizes, int n_in,
                              void* d_out, int out_size, void* d_ws, size_t ws_size,
                              hipStream_t stream) {
    (void)in_sizes; (void)n_in; (void)out_size; (void)ws_size;

    const float* enc    = (const float*)d_in[0];
    const float* target = (const float*)d_in[1];
    const float* proj_w = (const float*)d_in[2];
    const float* proj_b = (const float*)d_in[3];
    const float* w_ih0  = (const float*)d_in[4];
    const float* w_hh0  = (const float*)d_in[5];
    const float* b_ih0  = (const float*)d_in[6];
    const float* b_hh0  = (const float*)d_in[7];
    const float* w_ih1  = (const float*)d_in[8];
    const float* w_hh1  = (const float*)d_in[9];
    const float* b_ih1  = (const float*)d_in[10];
    const float* b_hh1  = (const float*)d_in[11];
    const float* lin_w  = (const float*)d_in[12];
    const float* lin_b  = (const float*)d_in[13];
    float* out = (float*)d_out;

    // Workspace carve-up (256B aligned); total ~99.5 MB
    char* ws = (char*)d_ws;
    size_t off = 0;
    auto alloc = [&](size_t bytes) -> void* {
        void* p = ws + off;
        off += (bytes + 255) & ~(size_t)255;
        return p;
    };
    unsigned short* linq   = (unsigned short*)alloc((size_t)V * H * 2);
    unsigned short* wih0q  = (unsigned short*)alloc((size_t)4 * H * E * 2);
    unsigned short* whh0q  = (unsigned short*)alloc((size_t)4 * H * H * 2);
    unsigned short* wih1q  = (unsigned short*)alloc((size_t)4 * H * H * 2);
    unsigned short* whh1q  = (unsigned short*)alloc((size_t)4 * H * H * 2);
    unsigned short* projq  = (unsigned short*)alloc((size_t)E * CATW * 2);
    unsigned short* catb   = (unsigned short*)alloc((size_t)B * T * CATW * 2);
    unsigned short* Hs     = (unsigned short*)alloc((size_t)B * T * H * 2);
    unsigned short* h0a    = (unsigned short*)alloc((size_t)B * H * 2);
    unsigned short* h0b    = (unsigned short*)alloc((size_t)B * H * 2);
    unsigned short* h1a    = (unsigned short*)alloc((size_t)B * H * 2);
    unsigned short* h1b    = (unsigned short*)alloc((size_t)B * H * 2);
    float*          c0     = (float*)alloc((size_t)B * H * 4);
    float*          c1     = (float*)alloc((size_t)B * H * 4);
    unsigned short* inp    = (unsigned short*)alloc((size_t)B * E * 2);

    auto conv = [&](const float* src, unsigned short* dst, size_t n) {
        k_f32_to_bf16<<<(unsigned)((n + 255) / 256), 256, 0, stream>>>(src, dst, n);
    };
    conv(lin_w,  linq,  (size_t)V * H);
    conv(w_ih0,  wih0q, (size_t)4 * H * E);
    conv(w_hh0,  whh0q, (size_t)4 * H * H);
    conv(w_ih1,  wih1q, (size_t)4 * H * H);
    conv(w_hh1,  whh1q, (size_t)4 * H * H);
    conv(proj_w, projq, (size_t)E * CATW);

    k_build_cat<<<(unsigned)(((size_t)B * T * CATW + 255) / 256), 256, 0, stream>>>(target, enc, catb);
    k_init_states<<<(B * H + 255) / 256, 256, 0, stream>>>(enc, h0a, h1a, c0, c1);

    for (int t = 0; t < T; ++t) {
        const unsigned short* h0p = (t & 1) ? h0b : h0a;
        unsigned short*       h0n = (t & 1) ? h0a : h0b;
        const unsigned short* h1p = (t & 1) ? h1b : h1a;
        unsigned short*       h1n = (t & 1) ? h1a : h1b;

        k_proj<<<E / 16, 32, 0, stream>>>(catb, t, projq, proj_b, inp);
        k_cell<<<H / 16, 32, 0, stream>>>(inp, E, wih0q, h0p, whh0q,
                                          b_ih0, b_hh0, c0, h0n,
                                          (unsigned short*)nullptr, t);
        k_cell<<<H / 16, 32, 0, stream>>>(h0n, H, wih1q, h1p, whh1q,
                                          b_ih1, b_hh1, c1, h1n, Hs, t);
    }

    k_gemm_logits<<<dim3(V / 256, (B * T) / 64), 256, 0, stream>>>(Hs, linq, lin_b, out);
    k_logsoftmax<<<B * T, 256, 0, stream>>>(out);
}
